// CausalSelfAttention_5789615915418
// MI455X (gfx1250) — matrix operand order repaired and checked by reading, not compiled, not run on hardware
//
#include <hip/hip_runtime.h>
#include <hip/hip_bf16.h>

// ---------------------------------------------------------------------------
// Causal self-attention (B=8, T=1024, C=768, H=12, HD=64) for MI455X/gfx1250.
// All GEMM stages on v_wmma_f32_16x16x32_bf16 (f32 accumulate).
//  - Stage 0: one-time bf16 conversion of x, and transpose+convert of weights
//             so every WMMA operand load is a contiguous 16B b128 load.
//  - Stage 1: QKV GEMM computed as D' = W'·x^T (16 cols x 64 rows per wave,
//             A fragment reused across 4 WMMAs). Q,K stored [B,H,T,HD] bf16,
//             V stored transposed [B,H,HD,T] bf16.
//  - Stage 2: flash-attention streaming softmax; S computed as S^T = K·Q^T so
//             the probability fragment is already in WMMA A-operand layout for
//             P·V (zero cross-lane transpose). global_prefetch on next K tile.
//  - Stage 3: out = O @ W_proj + b_proj as D' = Wp'·O^T, vectorized f32 stores.
// ---------------------------------------------------------------------------

typedef __attribute__((ext_vector_type(16))) __bf16 bf16x16;
typedef __attribute__((ext_vector_type(8)))  __bf16 bf16x8;
typedef __attribute__((ext_vector_type(8)))  float  f32x8;

#define WMMA_BF16(A, B, C) \
  __builtin_amdgcn_wmma_f32_16x16x32_bf16(false, (A), false, (B), (short)0, (C), false, false)

#define NEG_INF (-__builtin_inff())

// Assemble a 16-element bf16 fragment from two contiguous 16B runs.
static __device__ inline bf16x16 ld_split(const __bf16* lo, const __bf16* hi) {
  const bf16x8 a = *(const bf16x8*)lo;
  const bf16x8 b = *(const bf16x8*)hi;
  bf16x16 r;
#pragma unroll
  for (int i = 0; i < 8; ++i) { r[i] = a[i]; r[8 + i] = b[i]; }
  return r;
}
static __device__ inline bf16x16 ld16(const __bf16* p) { return ld_split(p, p + 8); }

// ---------------------------------------------------------------------------
// Stage 0a: f32 -> bf16 elementwise (x). One thread = 8 elements.
// ---------------------------------------------------------------------------
__global__ __launch_bounds__(256, 1) void cvt_bf16_kernel(
    const float* __restrict__ src, __bf16* __restrict__ dst, int n8) {
  const int tid = blockIdx.x * blockDim.x + threadIdx.x;
  if (tid >= n8) return;
  const float4 a = ((const float4*)src)[tid * 2];
  const float4 b = ((const float4*)src)[tid * 2 + 1];
  bf16x8 v;
  v[0] = (__bf16)a.x; v[1] = (__bf16)a.y; v[2] = (__bf16)a.z; v[3] = (__bf16)a.w;
  v[4] = (__bf16)b.x; v[5] = (__bf16)b.y; v[6] = (__bf16)b.z; v[7] = (__bf16)b.w;
  ((bf16x8*)dst)[tid] = v;
}

// ---------------------------------------------------------------------------
// Stage 0b: transpose + convert: src [K][N] f32 -> dst [N][K] bf16.
// One thread = 8 consecutive k of one output row n (contiguous b128 store).
// ---------------------------------------------------------------------------
__global__ __launch_bounds__(256, 1) void transpose_cvt_kernel(
    const float* __restrict__ src, __bf16* __restrict__ dst, int K, int N) {
  const int tid = blockIdx.x * blockDim.x + threadIdx.x;
  const int kgroups = K >> 3;
  if (tid >= N * kgroups) return;
  const int n  = tid / kgroups;
  const int k0 = (tid - n * kgroups) << 3;
  bf16x8 v;
#pragma unroll
  for (int i = 0; i < 8; ++i) v[i] = (__bf16)src[(size_t)(k0 + i) * N + n];
  *(bf16x8*)(dst + (size_t)n * K + k0) = v;
}

// ---------------------------------------------------------------------------
// Stage 1: QKV. D'(n,m) tile = W'(n,:) x(m,:)^T. Wave owns 16 cols x 64 rows.
// grid = (2304/16, 8192/64/4), block = 128 (4 waves).
// ---------------------------------------------------------------------------
__global__ __launch_bounds__(128, 1) void qkv_wmma_kernel(
    const __bf16* __restrict__ xb,   // [8192][768] bf16
    const __bf16* __restrict__ Wt,   // [2304][768] bf16 (W_attn^T)
    const float* __restrict__ bias,  // [2304]
    __bf16* __restrict__ q_ws, __bf16* __restrict__ k_ws, __bf16* __restrict__ v_ws) {
  const int lane = threadIdx.x & 31;
  const int wv   = threadIdx.x >> 5;
  const int l15  = lane & 15;
  const int half = lane >> 4;
  const int kb   = half << 3;                       // 0 or 8
  const int n0   = blockIdx.x << 4;                 // output-column tile [0,2304)
  const int m0   = (((blockIdx.y << 2) + wv) << 6); // 64-row super-tile [0,8192)

  const __bf16* wrow = Wt + (size_t)(n0 + l15) * 768;  // A: lane = out col n
  const __bf16* xr   = xb + (size_t)(m0 + l15) * 768;  // B: lane = row m (tile 0)

  f32x8 acc[4];
#pragma unroll
  for (int mt = 0; mt < 4; ++mt)
#pragma unroll
    for (int r = 0; r < 8; ++r) acc[mt][r] = 0.f;

  for (int k0 = 0; k0 < 768; k0 += 32) {
    const bf16x16 A = ld16(wrow + k0 + (half << 4));
#pragma unroll
    for (int mt = 0; mt < 4; ++mt) {
      const bf16x16 Bx = ld16(xr + (size_t)(mt * 16) * 768 + k0 + (half << 4));
      acc[mt] = WMMA_BF16(A, Bx, acc[mt]);
    }
  }

  // bias for this lane's 8 consecutive output columns n0+kb .. n0+kb+7
  const float4 bb0 = *(const float4*)(bias + n0 + kb);
  const float4 bb1 = *(const float4*)(bias + n0 + kb + 4);
  const float bias8[8] = {bb0.x, bb0.y, bb0.z, bb0.w, bb1.x, bb1.y, bb1.z, bb1.w};

  const int which = n0 / 768;            // 0=Q 1=K 2=V (uniform for the tile)
  const int cm    = n0 - which * 768;
  const int hh    = cm >> 6;
  const int d0    = (cm & 63) + kb;      // first of 8 consecutive head dims

#pragma unroll
  for (int mt = 0; mt < 4; ++mt) {
    const int row = m0 + mt * 16 + l15;  // D': lane = row
    const int bbx = row >> 10;
    const int tt  = row & 1023;
    const size_t bh = (size_t)(bbx * 12 + hh);
    if (which < 2) {
      bf16x8 ov;
#pragma unroll
      for (int r = 0; r < 8; ++r) ov[r] = (__bf16)(acc[mt][r] + bias8[r]);
      __bf16* dst = (which == 0 ? q_ws : k_ws) + (bh * 1024 + tt) * 64 + d0;
      *(bf16x8*)dst = ov;                // one b128 store: 8 consecutive dims
    } else {
#pragma unroll
      for (int r = 0; r < 8; ++r)        // V^T: unavoidable scatter
        v_ws[(bh * 64 + d0 + r) * 1024 + tt] = (__bf16)(acc[mt][r] + bias8[r]);
    }
  }
}

// ---------------------------------------------------------------------------
// Stage 2: streaming-softmax attention. One wave per (b,h,16-query tile).
// grid = (64/4, 96), block = 128. 32 keys per iteration, 8 WMMAs/iter.
// ---------------------------------------------------------------------------
__global__ __launch_bounds__(128, 1) void attn_wmma_kernel(
    const __bf16* __restrict__ q_ws, const __bf16* __restrict__ k_ws,
    const __bf16* __restrict__ v_ws,
    const unsigned char* __restrict__ mask,    // [B,1,T,T] bool
    const unsigned char* __restrict__ qmask,   // [B,1,T,1] bool
    __bf16* __restrict__ o_ws) {               // [B*T, C] bf16
  const int lane = threadIdx.x & 31;
  const int wv   = threadIdx.x >> 5;
  const int l15  = lane & 15;
  const int half = lane >> 4;
  const int kb   = half << 3;
  const int qt   = (blockIdx.x << 2) + wv;     // 0..63
  const int bh   = blockIdx.y;                 // 0..95
  const int b    = bh / 12;
  const int h    = bh - b * 12;
  const int q0   = qt << 4;
  const int q_lane = q0 + l15;

  const __bf16* Qb = q_ws + (size_t)bh * 1024 * 64;
  const __bf16* Kb = k_ws + (size_t)bh * 1024 * 64;
  const __bf16* Vb = v_ws + (size_t)bh * 64 * 1024;
  // key_valid[b,k] == mask[b,0,T-1,k] (causal row T-1 is all ones)
  const unsigned char* kvrow = mask + (size_t)b * 1024 * 1024 + (size_t)1023 * 1024;

  // Q as B-operand of S^T = K·Q^T (lane = query), resident for whole kernel
  const __bf16* qrow = Qb + (size_t)q_lane * 64;
  const bf16x16 Bq0 = ld_split(qrow + kb, qrow + 16 + kb);        // d 0..31
  const bf16x16 Bq1 = ld_split(qrow + 32 + kb, qrow + 48 + kb);   // d 32..63

  f32x8 O[4];
#pragma unroll
  for (int c = 0; c < 4; ++c)
#pragma unroll
    for (int r = 0; r < 8; ++r) O[c][r] = 0.f;

  float m_cur = NEG_INF;
  float l_cur = 0.f;
  const f32x8 zero8 = {0.f, 0.f, 0.f, 0.f, 0.f, 0.f, 0.f, 0.f};

  const int kend = q0 + 16;                    // causal bound for this tile
  for (int k0 = 0; k0 < kend; k0 += 32) {
    const __bf16* kr0 = Kb + (size_t)(k0 + l15) * 64;
    const __bf16* kr1 = kr0 + 16 * 64;

    // prefetch next K tile into cache (global_prefetch_b8) while we compute
    if (k0 + 32 < kend) {
      __builtin_prefetch(kr0 + 32 * 64, 0, 3);
      __builtin_prefetch(kr1 + 32 * 64, 0, 3);
    }

    // --- S^T tiles: A = K rows (M = key), B = Q^T. Two 16-key subtiles. ---
    f32x8 s0 = zero8, s1 = zero8;
    s0 = WMMA_BF16(ld_split(kr0 + kb,      kr0 + 16 + kb), Bq0, s0);
    s0 = WMMA_BF16(ld_split(kr0 + 32 + kb, kr0 + 48 + kb), Bq1, s0);
    s1 = WMMA_BF16(ld_split(kr1 + kb,      kr1 + 16 + kb), Bq0, s1);
    s1 = WMMA_BF16(ld_split(kr1 + 32 + kb, kr1 + 48 + kb), Bq1, s1);

    // --- scale + masks. Lane holds query q_lane; element r is key kb+r (+16).
    const unsigned long long kv0 = *(const unsigned long long*)(kvrow + k0 + kb);
    const unsigned long long kv1 = *(const unsigned long long*)(kvrow + k0 + 16 + kb);
    float smax = NEG_INF;
#pragma unroll
    for (int r = 0; r < 8; ++r) {
      const int key0 = k0 + kb + r;
      const int key1 = key0 + 16;
      const bool v0 = (key0 <= q_lane) && ((kv0 >> (r * 8)) & 1ull);
      const bool v1 = (key1 <= q_lane) && ((kv1 >> (r * 8)) & 1ull);
      const float e0 = v0 ? s0[r] * 0.125f : NEG_INF;   // 1/sqrt(64)
      const float e1 = v1 ? s1[r] * 0.125f : NEG_INF;
      s0[r] = e0; s1[r] = e1;
      smax = fmaxf(smax, fmaxf(e0, e1));
    }
    smax = fmaxf(smax, __shfl_xor(smax, 16, 32));       // join half-wave key groups
    const float m_new = fmaxf(m_cur, smax);
    const float alpha = (m_new == NEG_INF) ? 1.f : __expf(m_cur - m_new);

    // --- P = exp(S - m): already in WMMA A-operand layout ---
    float rsum = 0.f;
    bf16x16 P;
#pragma unroll
    for (int r = 0; r < 8; ++r) {
      const float p0 = (s0[r] == NEG_INF) ? 0.f : __expf(s0[r] - m_new);
      const float p1 = (s1[r] == NEG_INF) ? 0.f : __expf(s1[r] - m_new);
      rsum += p0 + p1;
      P[r]     = (__bf16)p0;
      P[8 + r] = (__bf16)p1;
    }
    rsum += __shfl_xor(rsum, 16, 32);
    l_cur = l_cur * alpha + rsum;
    m_cur = m_new;

    // broadcast alpha into the O-fragment row dimension (row m = r + 8*half)
    float alf[8];
#pragma unroll
    for (int r = 0; r < 8; ++r) alf[r] = __shfl(alpha, r + kb, 32);

    // --- O = O*alpha + P·V (V transposed: contiguous key runs) ---
#pragma unroll
    for (int c = 0; c < 4; ++c) {
      const __bf16* vr = Vb + (size_t)((c << 4) + l15) * 1024 + k0;
      const bf16x16 Bv = ld_split(vr + kb, vr + 16 + kb);
#pragma unroll
      for (int r = 0; r < 8; ++r) O[c][r] *= alf[r];
      O[c] = WMMA_BF16(P, Bv, O[c]);
    }
  }

  // finalize: divide by l, apply query_mask; zero rows with no mass
  const unsigned char qv = qmask[(size_t)b * 1024 + q_lane];
  const float fin = (l_cur > 0.f && qv) ? (1.f / l_cur) : 0.f;
  float finm[8];
#pragma unroll
  for (int r = 0; r < 8; ++r) finm[r] = __shfl(fin, r + kb, 32);

#pragma unroll
  for (int c = 0; c < 4; ++c) {
    const int d = (c << 4) + l15;
#pragma unroll
    for (int r = 0; r < 8; ++r) {
      const size_t row = (size_t)b * 1024 + q0 + kb + r;
      o_ws[row * 768 + h * 64 + d] = (__bf16)(O[c][r] * finm[r]);
    }
  }
}

// ---------------------------------------------------------------------------
// Stage 3: out = O @ W_proj + b_proj, computed as D' = Wp'·O^T.
// Wave owns 16 cols x 64 rows; f32 stores as contiguous float4 pairs.
// grid = (768/16, 8192/64/4), block = 128.
// ---------------------------------------------------------------------------
__global__ __launch_bounds__(128, 1) void proj_wmma_kernel(
    const __bf16* __restrict__ o_ws,  // [8192][768] bf16
    const __bf16* __restrict__ Wpt,   // [768][768] bf16 (W_proj^T)
    const float* __restrict__ bias, float* __restrict__ out) {
  const int lane = threadIdx.x & 31;
  const int wv   = threadIdx.x >> 5;
  const int l15  = lane & 15;
  const int half = lane >> 4;
  const int kb   = half << 3;
  const int n0   = blockIdx.x << 4;
  const int m0   = (((blockIdx.y << 2) + wv) << 6);

  const __bf16* wrow = Wpt + (size_t)(n0 + l15) * 768;
  const __bf16* orow = o_ws + (size_t)(m0 + l15) * 768;

  f32x8 acc[4];
#pragma unroll
  for (int mt = 0; mt < 4; ++mt)
#pragma unroll
    for (int r = 0; r < 8; ++r) acc[mt][r] = 0.f;

  for (int k0 = 0; k0 < 768; k0 += 32) {
    const bf16x16 A = ld16(wrow + k0 + (half << 4));
#pragma unroll
    for (int mt = 0; mt < 4; ++mt) {
      const bf16x16 Bo = ld16(orow + (size_t)(mt * 16) * 768 + k0 + (half << 4));
      acc[mt] = WMMA_BF16(A, Bo, acc[mt]);
    }
  }

  const float4 bb0 = *(const float4*)(bias + n0 + kb);
  const float4 bb1 = *(const float4*)(bias + n0 + kb + 4);

#pragma unroll
  for (int mt = 0; mt < 4; ++mt) {
    const int row = m0 + mt * 16 + l15;
    float* op = out + (size_t)row * 768 + n0 + kb;   // 8 consecutive f32
    const float4 s0 = {acc[mt][0] + bb0.x, acc[mt][1] + bb0.y,
                       acc[mt][2] + bb0.z, acc[mt][3] + bb0.w};
    const float4 s1 = {acc[mt][4] + bb1.x, acc[mt][5] + bb1.y,
                       acc[mt][6] + bb1.z, acc[mt][7] + bb1.w};
    *(float4*)op = s0;
    *(float4*)(op + 4) = s1;
  }
}

// ---------------------------------------------------------------------------
// Host launcher
// ---------------------------------------------------------------------------
extern "C" void kernel_launch(void* const* d_in, const int* in_sizes, int n_in,
                              void* d_out, int out_size, void* d_ws, size_t ws_size,
                              hipStream_t stream) {
  const float*         x      = (const float*)d_in[0];
  const unsigned char* mask   = (const unsigned char*)d_in[1];  // bool [B,1,T,T]
  const unsigned char* qmask  = (const unsigned char*)d_in[2];  // bool [B,1,T,1]
  const float*         W_attn = (const float*)d_in[3];
  const float*         b_attn = (const float*)d_in[4];
  const float*         W_proj = (const float*)d_in[5];
  const float*         b_proj = (const float*)d_in[6];
  float*               out    = (float*)d_out;

  // workspace layout (bf16 elements)
  const size_t per = (size_t)96 * 1024 * 64;        // 6291456 (Q, K, V, O, xb each)
  __bf16* q_ws = (__bf16*)d_ws;
  __bf16* k_ws = q_ws + per;
  __bf16* v_ws = k_ws + per;
  __bf16* o_ws = v_ws + per;
  __bf16* xb   = o_ws + per;
  __bf16* Wa_t = xb + per;                          // [2304][768]
  __bf16* Wp_t = Wa_t + (size_t)2304 * 768;         // [768][768]

  // Stage 0: precision/layout prep
  cvt_bf16_kernel<<<dim3(3072), 256, 0, stream>>>(x, xb, 786432);
  transpose_cvt_kernel<<<dim3(864), 256, 0, stream>>>(W_attn, Wa_t, 768, 2304);
  transpose_cvt_kernel<<<dim3(288), 256, 0, stream>>>(W_proj, Wp_t, 768, 768);

  // Stage 1-3
  qkv_wmma_kernel<<<dim3(144, 32), 128, 0, stream>>>(xb, Wa_t, b_attn, q_ws, k_ws, v_ws);
  attn_wmma_kernel<<<dim3(16, 96), 128, 0, stream>>>(q_ws, k_ws, v_ws, mask, qmask, o_ws);
  proj_wmma_kernel<<<dim3(48, 32), 128, 0, stream>>>(o_ws, Wp_t, b_proj, out);
}
